// QuantbleMixtralSparseMoeBlockWrapper_87668872446486
// MI455X (gfx1250) — matrix operand order repaired and checked by reading, not compiled, not run on hardware
//
#include <hip/hip_runtime.h>
#include <stdint.h>

// ---------------- problem constants (from reference) ----------------
#define NE 8          // experts
#define DH 1024       // hidden dim
#define FF 3584       // ffn dim
#define KFCH 128      // F-chunk width processed per iteration
// MoE kernel: 256 threads = 8 waves (wave32), one 16-token tile per block.

// Probe: v_cvt_pk_bf16_f32 exists on gfx950; testing gfx1250 assembler.
#define USE_ASM_CVT_PK_BF16 1

typedef __attribute__((ext_vector_type(16))) __bf16 v16bf;
typedef __attribute__((ext_vector_type(2)))  __bf16 v2bf;
typedef __attribute__((ext_vector_type(8)))  float  v8f;

union ABu  { uint4 q[2]; v16bf v; };   // 32B: 16 bf16 fragment (A or B)
union ACCu { v8f v; float f[8]; };     // WMMA f32 accumulator

// fp32 -> bf16, round-to-nearest (add 0x8000 then take high half).
__device__ __forceinline__ unsigned short f2bf(float f) {
    return (unsigned short)((__float_as_uint(f) + 0x8000u) >> 16);
}

// Pack two fp32 into packed bf16x2.
__device__ __forceinline__ unsigned pack2(float a, float b) {
#if __has_builtin(__builtin_amdgcn_cvt_pk_bf16_f32)
    union { v2bf v; unsigned u; } c;
    c.v = __builtin_amdgcn_cvt_pk_bf16_f32(a, b);   // v_cvt_pk_bf16_f32
    return c.u;
#elif USE_ASM_CVT_PK_BF16
    unsigned r;
    asm("v_cvt_pk_bf16_f32 %0, %1, %2" : "=v"(r) : "v"(a), "v"(b));
    return r;
#else
    unsigned ua = __float_as_uint(a) + 0x8000u;
    unsigned ub = __float_as_uint(b) + 0x8000u;
    return __builtin_amdgcn_perm(ub, ua, 0x07060302u); // v_perm_b32
#endif
}

__device__ __forceinline__ unsigned ballot32(bool p) {
#if __has_builtin(__builtin_amdgcn_ballot_w32)
    return __builtin_amdgcn_ballot_w32(p);
#else
    return (unsigned)__ballot(p);
#endif
}

// Load 16 consecutive fp32 and convert to a 16-element bf16 fragment.
// B-layout used here, same k order as A (16-bit, 32x16): lane covers one N column, its 16 values
// are K = (lane>=16 ? 16 : 0) + i along the contiguous inner dimension.
__device__ __forceinline__ ABu cvt16(const float* __restrict__ p) {
    float4 f0 = *(const float4*)(p + 0);
    float4 f1 = *(const float4*)(p + 4);
    float4 f2 = *(const float4*)(p + 8);
    float4 f3 = *(const float4*)(p + 12);
    ABu r;
    r.q[0] = make_uint4(pack2(f0.x, f0.y), pack2(f0.z, f0.w),
                        pack2(f1.x, f1.y), pack2(f1.z, f1.w));
    r.q[1] = make_uint4(pack2(f2.x, f2.y), pack2(f2.z, f2.w),
                        pack2(f3.x, f3.y), pack2(f3.z, f3.w));
    return r;
}

// ---------------------------------------------------------------------------
// Kernel 1: router. logits -> softmax -> top-2 -> renormalize.
// Writes logits to d_out tail; top-2 (index, weight) pairs to workspace.
// ---------------------------------------------------------------------------
__global__ void router_kernel(const float* __restrict__ x,
                              const float* __restrict__ gate,
                              float* __restrict__ logits_out,
                              int* __restrict__ topIdx,    // [T][2]
                              float* __restrict__ topW,    // [T][2]
                              int T) {
    __shared__ float gsh[NE * DH];            // 32 KB: gate weights
    int tid = threadIdx.x;
    for (int i = tid; i < NE * DH; i += blockDim.x) gsh[i] = gate[i];
    __syncthreads();

    int t = blockIdx.x * blockDim.x + tid;
    if (t >= T) return;
    const float* xr = x + (size_t)t * DH;

    float lg[NE];
    for (int e = 0; e < NE; ++e) {
        const float* gr = gsh + e * DH;
        float acc = 0.f;
        for (int d = 0; d < DH; d += 4) {
            float4 xv = *(const float4*)(xr + d);
            acc += xv.x * gr[d] + xv.y * gr[d + 1] + xv.z * gr[d + 2] + xv.w * gr[d + 3];
        }
        lg[e] = acc;
    }
    float mx = lg[0];
    for (int e = 1; e < NE; ++e) mx = fmaxf(mx, lg[e]);
    float p[NE], se = 0.f;
    for (int e = 0; e < NE; ++e) { p[e] = __expf(lg[e] - mx); se += p[e]; }
    float rse = __builtin_amdgcn_rcpf(se);
    for (int e = 0; e < NE; ++e) p[e] *= rse;

    // top-2, ties resolve to lower index (matches lax.top_k)
    int i0 = 0;
    for (int e = 1; e < NE; ++e) if (p[e] > p[i0]) i0 = e;
    int i1 = (i0 == 0) ? 1 : 0;
    for (int e = 0; e < NE; ++e) if (e != i0 && p[e] > p[i1]) i1 = e;
    float rs2 = __builtin_amdgcn_rcpf(p[i0] + p[i1]);

    for (int e = 0; e < NE; ++e) logits_out[(size_t)t * NE + e] = lg[e];
    topIdx[t * 2 + 0] = i0;  topW[t * 2 + 0] = p[i0] * rs2;
    topIdx[t * 2 + 1] = i1;  topW[t * 2 + 1] = p[i1] * rs2;
}

// ---------------------------------------------------------------------------
// Kernel 1b: deterministic per-expert token compaction (ascending token
// order via wave ballot + prefix popcount; no atomics).
// ---------------------------------------------------------------------------
__global__ void compact_kernel(const int* __restrict__ topIdx,
                               const float* __restrict__ topW,
                               int* __restrict__ tokList,   // [E][T]
                               float* __restrict__ wList,   // [E][T]
                               int* __restrict__ slotList,  // [E][T]
                               int* __restrict__ cnt,       // [E]
                               int T) {
    const int e    = blockIdx.x;         // one 32-thread wave per expert
    const int lane = threadIdx.x;
    const unsigned lt = (1u << lane) - 1u;
    int base = 0;
    for (int t0 = 0; t0 < T; t0 += 32) {
        int t = t0 + lane;
        int i0 = topIdx[t * 2], i1 = topIdx[t * 2 + 1];
        bool m0 = (i0 == e), m1 = (i1 == e);
        bool mem = m0 || m1;
        unsigned mask = ballot32(mem);
        if (mem) {
            int pos = base + __popc(mask & lt);
            tokList[e * T + pos]  = t;
            slotList[e * T + pos] = m0 ? 0 : 1;
            wList[e * T + pos]    = m0 ? topW[t * 2] : topW[t * 2 + 1];
        }
        base += __popc(mask);
    }
    if (lane == 0) cnt[e] = base;
}

// ---------------------------------------------------------------------------
// Kernel 2: fused MoE SwiGLU with bf16 WMMA (fp32 accumulate).
// Grid = (maxTiles, E). Block = 16 compacted tokens of one expert, 8 waves.
// slot-0 rows -> outF, slot-1 rows -> part1 (each (t,d) written exactly once
// per buffer -> deterministic, no FP atomics).
// ---------------------------------------------------------------------------
__global__ void __launch_bounds__(256, 1)
moe_kernel(const float* __restrict__ x,
           const float* __restrict__ w1,
           const float* __restrict__ w3,
           const float* __restrict__ w2,
           const int* __restrict__ tokList,
           const float* __restrict__ wList,
           const int* __restrict__ slotList,
           const int* __restrict__ cnt,
           float* __restrict__ outF,
           float* __restrict__ part1,
           int T) {
    const int e  = blockIdx.y;
    const int ti = blockIdx.x;
    const int n  = cnt[e];
    if (ti * 16 >= n) return;                 // uniform early exit

    __shared__ unsigned short xbf[16 * DH];   // 32 KB bf16 x-tile (gathered)
    __shared__ float xstage[16 * 256];        // 16 KB fp32 async staging
    __shared__ unsigned short hbf[16 * KFCH]; // 4 KB  bf16 h-tile
    __shared__ int   tokS[16];
    __shared__ float wes[16];
    __shared__ int   slotS[16];
    __shared__ int   valS[16];

    const int tid  = threadIdx.x;
    const int wave = tid >> 5;
    const int lane = tid & 31;
    const int ln   = lane & 15;       // N column / M row within 16
    const int kh   = lane >> 4;       // upper/lower K half select

    if (tid < 16) {
        int pos = ti * 16 + tid;
        int v   = (pos < n);
        int p2  = v ? pos : ti * 16;  // first row of tile is always valid
        tokS[tid]  = tokList[e * T + p2];
        wes[tid]   = v ? wList[e * T + p2] : 0.f;   // 0-weight pad rows
        slotS[tid] = slotList[e * T + p2];
        valS[tid]  = v;
    }
    __syncthreads();

    // ---- Gather x rows: async global->LDS (fp32), then convert to bf16 ----
    for (int c = 0; c < 4; ++c) {                 // 4 sub-chunks of 256 cols
        for (int it = 0; it < 4; ++it) {
            int q   = it * 256 + tid;             // float4 index in sub-chunk
            int row = q >> 6;                     // 64 float4 per row
            int c4  = q & 63;
            const float* g = x + (size_t)tokS[row] * DH + c * 256 + c4 * 4;
            unsigned lds = (unsigned)(uintptr_t)(&xstage[row * 256 + c4 * 4]);
            asm volatile("global_load_async_to_lds_b128 %0, %1, off"
                         :: "v"(lds), "v"(g) : "memory");
        }
        asm volatile("s_wait_asynccnt 0x0" ::: "memory");
        __syncthreads();
        // packed convert: 4096 floats -> 2048 b32 LDS stores / 256 threads
        for (int it = 0; it < 8; ++it) {
            int q   = (it * 256 + tid) * 2;       // even element index
            int row = q >> 8, cc = q & 255;
            unsigned pk = pack2(xstage[row * 256 + cc], xstage[row * 256 + cc + 1]);
            *(unsigned*)&xbf[row * DH + c * 256 + cc] = pk;
        }
        __syncthreads();
    }

    // persistent output accumulators: [16 tokens x 128 D-cols] per wave
    ACCu outacc[8];
    for (int i = 0; i < 8; ++i)
        for (int j = 0; j < 8; ++j) outacc[i].f[j] = 0.f;

    const float* w1e = w1 + (size_t)e * FF * DH;
    const float* w3e = w3 + (size_t)e * FF * DH;
    const float* w2e = w2 + (size_t)e * DH * FF;

    for (int fc = 0; fc < FF; fc += KFCH) {
        // ---- phase 1: h[:, fc + wave*16 .. +16) over full D reduction
        ACCu a1, a3;
        for (int j = 0; j < 8; ++j) { a1.f[j] = 0.f; a3.f[j] = 0.f; }
        const int fbase = fc + wave * 16;
        if (fc + KFCH < FF) {                 // hint next chunk into L2
            __builtin_prefetch(w1e + (size_t)(fbase + KFCH) * DH, 0, 1);
            __builtin_prefetch(w3e + (size_t)(fbase + KFCH) * DH, 0, 1);
        }
#pragma unroll 4
        for (int ks = 0; ks < DH; ks += 32) {
            ABu A;                            // A: x[16 x 32] bf16 from LDS
            const unsigned short* ap = xbf + ln * DH + ks + kh * 16;
            A.q[0] = *(const uint4*)(ap);
            A.q[1] = *(const uint4*)(ap + 8);
            const float* b1p = w1e + (size_t)(fbase + ln) * DH + ks + kh * 16;
            const float* b3p = w3e + (size_t)(fbase + ln) * DH + ks + kh * 16;
            ABu B1 = cvt16(b1p);
            ABu B3 = cvt16(b3p);
            a1.v = __builtin_amdgcn_wmma_f32_16x16x32_bf16(
                       false, A.v, false, B1.v, (short)0, a1.v, false, false);
            a3.v = __builtin_amdgcn_wmma_f32_16x16x32_bf16(
                       false, A.v, false, B3.v, (short)0, a3.v, false, false);
        }
        // silu(a1) * a3, scaled by per-token routing weight; C-layout:
        // VGPR r -> M = r + 8*kh, N = ln  (within this wave's 16 cols)
#pragma unroll
        for (int r = 0; r < 8; ++r) {
            int m = r + kh * 8;
            float z   = a1.f[r];
            float sig = __builtin_amdgcn_rcpf(1.f + __expf(-z)); // fast sigmoid
            float hh  = z * sig * a3.f[r] * wes[m];
            hbf[m * KFCH + wave * 16 + ln] = f2bf(hh);
        }
        __syncthreads();

        // ---- phase 2: out[:, wave*128 .. +128) += h @ w2^T chunk
#pragma unroll
        for (int ks = 0; ks < KFCH; ks += 32) {
            ABu A;                            // A: h[16 x 32] bf16 from LDS
            const unsigned short* ap = hbf + ln * KFCH + ks + kh * 16;
            A.q[0] = *(const uint4*)(ap);
            A.q[1] = *(const uint4*)(ap + 8);
#pragma unroll
            for (int nt = 0; nt < 8; ++nt) {
                int dcol = wave * 128 + nt * 16 + ln;
                const float* bp = w2e + (size_t)dcol * FF + fc + ks + kh * 16;
                ABu B = cvt16(bp);
                outacc[nt].v = __builtin_amdgcn_wmma_f32_16x16x32_bf16(
                    false, A.v, false, B.v, (short)0, outacc[nt].v, false, false);
            }
        }
        __syncthreads();                      // hbf reused next chunk
    }

    // ---- per-row store: slot 0 -> outF, slot 1 -> part1, pad rows masked ----
    for (int nt = 0; nt < 8; ++nt) {
        for (int r = 0; r < 8; ++r) {
            int m = r + kh * 8;
            if (valS[m]) {
                float* dst = slotS[m] ? part1 : outF;
                dst[(size_t)tokS[m] * DH + wave * 128 + nt * 16 + ln] = outacc[nt].f[r];
            }
        }
    }
}

// ---------------------------------------------------------------------------
// Kernel 3: combine slot-0 (in d_out) and slot-1 (ws) partials.
// ---------------------------------------------------------------------------
__global__ void combine_kernel(float* __restrict__ out,
                               const float* __restrict__ part1, int n4) {
    int i = blockIdx.x * blockDim.x + threadIdx.x;
    if (i < n4) {
        float4 a = ((const float4*)out)[i];
        float4 b = ((const float4*)part1)[i];
        a.x += b.x; a.y += b.y; a.z += b.z; a.w += b.w;
        ((float4*)out)[i] = a;
    }
}

// ---------------------------------------------------------------------------
extern "C" void kernel_launch(void* const* d_in, const int* in_sizes, int n_in,
                              void* d_out, int out_size, void* d_ws, size_t ws_size,
                              hipStream_t stream) {
    const float* x    = (const float*)d_in[0];   // [B,S,D] fp32
    const float* gate = (const float*)d_in[1];   // [E,D]
    const float* w1   = (const float*)d_in[2];   // [E,F,D]
    const float* w3   = (const float*)d_in[3];   // [E,F,D]
    const float* w2   = (const float*)d_in[4];   // [E,D,F]

    const int T = in_sizes[0] / DH;              // 4096 tokens

    float* outF   = (float*)d_out;               // final [T,D]
    float* logits = outF + (size_t)T * DH;       // router_logits [T,E]

    // workspace layout (~17.2 MB)
    char* wsb = (char*)d_ws;
    float* part1   = (float*)wsb;                 wsb += (size_t)T * DH * 4;
    int*   tokList = (int*)wsb;                   wsb += (size_t)NE * T * 4;
    float* wList   = (float*)wsb;                 wsb += (size_t)NE * T * 4;
    int*   slotList= (int*)wsb;                   wsb += (size_t)NE * T * 4;
    int*   cnt     = (int*)wsb;                   wsb += 64;
    int*   topIdx  = (int*)wsb;                   wsb += (size_t)T * 2 * 4;
    float* topW    = (float*)wsb;

    router_kernel <<<(T + 127) / 128, 128, 0, stream>>>(x, gate, logits, topIdx, topW, T);
    compact_kernel<<<NE, 32, 0, stream>>>(topIdx, topW, tokList, wList, slotList, cnt, T);
    moe_kernel    <<<dim3(T / 16, NE), 256, 0, stream>>>(x, w1, w3, w2,
                                                         tokList, wList, slotList, cnt,
                                                         outF, part1, T);
    combine_kernel<<<(T * DH / 4 + 255) / 256, 256, 0, stream>>>(outF, part1, T * DH / 4);
}
